// GraphUNet_11312943858316
// MI455X (gfx1250) — matrix operand order repaired and checked by reading, not compiled, not run on hardware
//
#include <hip/hip_runtime.h>
#include <hip/hip_bf16.h>

// ---------------- problem constants (match reference) ----------------
#define B_    4
#define V_    20000
#define E_    320000
#define IN_   64
#define COND_ 32
#define ENC_  128
#define M_    (B_ * V_)        // 80000 rows = 625 tiles of 128

typedef __bf16 bf16_t;
typedef __attribute__((ext_vector_type(16))) __bf16 v16bf;
typedef __attribute__((ext_vector_type(8)))  __bf16 v8bf;
typedef __attribute__((ext_vector_type(8)))  float  v8f;

// ---------------- WMMA GEMM: C[M,N] = concat(A1,A2)[M,K] @ W[K,N] ----------------
// 256 threads = 8 waves; block tile 128x64, K-step 32. Waves arranged 4(M)x2(N),
// each wave owns a 32x32 C tile -> 2 A-frags x 2 B-frags -> 4 v_wmma per k-step.
// A2 = optional second column-block source (cond_x broadcast over batch via
// row % V_ [compile-time-constant modulo, hoisted], or a skip buffer).
#define BM  128
#define BN  64
#define BK  32
#define ALD 40   // 80B rows -> 20-dword bank stride, conflict-free b128 frag loads
#define BLD 40

__global__ __launch_bounds__(256) void k_gemm_bf16(
    const float* __restrict__ A1, int lda1, int k1,
    const float* __restrict__ A2, int lda2, int a2_bcast,  // 1: A2 row = row % V_
    const float* __restrict__ W,                 // [K, N] row-major
    const float* __restrict__ bias,              // [N] or null
    const float* __restrict__ addsrc,            // [M, N] residual or null
    float* __restrict__ C, int N, int K, int relu)
{
  __shared__ bf16_t As[BM * ALD];
  __shared__ bf16_t Bs[BN * BLD];                // stored transposed: [n][k]
  const int tid  = threadIdx.x;
  const int lane = tid & 31;
  const int wave = tid >> 5;
  const int wm   = (wave >> 1) << 5;             // 0,32,64,96
  const int wn   = (wave & 1) << 5;              // 0,32
  const int m0   = blockIdx.x * BM;
  const int n0   = blockIdx.y * BN;

  // A staging: 2 threads per row, 16 consecutive k each (128x32 tile)
  const int arow = tid >> 1;                     // 0..127
  const int acol = (tid & 1) << 4;               // 0,16
  const int grow = m0 + arow;
  const float* __restrict__ ap1 = A1 + (size_t)grow * lda1;
  const float* __restrict__ ap2 =
      A2 ? (A2 + (size_t)(a2_bcast ? (grow % V_) : grow) * lda2) : nullptr;
  // B staging: 4 threads per n-row, 8 consecutive k each (64x32 tile, transposed)
  const int brow = tid >> 2;                     // 0..63
  const int bcol = (tid & 3) << 3;               // 0,8,16,24
  const float* __restrict__ wp = W + (n0 + brow);

  v8f acc00 = {}, acc01 = {}, acc10 = {}, acc11 = {};

  for (int kt = 0; kt < K; kt += BK) {
    if (kt + BK < K)                             // gfx1250 global_prefetch_b8
      __builtin_prefetch(&wp[(size_t)(kt + BK) * N], 0, 0);
    // stage A tile f32 -> bf16 (native packed bf16 convert)
    #pragma unroll
    for (int j = 0; j < 16; ++j) {
      int k = kt + acol + j;
      float v = (k < k1) ? ap1[k] : ap2[k - k1];
      As[arow * ALD + acol + j] = (bf16_t)v;
    }
    // stage W tile transposed: Bs[n][k]
    #pragma unroll
    for (int j = 0; j < 8; ++j) {
      Bs[brow * BLD + bcol + j] = (bf16_t)wp[(size_t)(kt + bcol + j) * N];
    }
    __syncthreads();
    {
      // A fragment: ISA 16-bit A 16x32 layout — lanes<16 hold K {0..7,16..23},
      // lanes>=16 hold K {8..15,24..31}, M = lane&15.
      const int ml = lane & 15;
      const int hk = (lane >> 4) << 3;           // 0 / 8
      const int kb = (lane >> 4) << 3;           // 0 / 8
      v8bf a0l = *(const v8bf*)&As[(wm + ml) * ALD + hk];
      v8bf a0h = *(const v8bf*)&As[(wm + ml) * ALD + hk + 16];
      v16bf a0 = __builtin_shufflevector(a0l, a0h,
          0,1,2,3,4,5,6,7,8,9,10,11,12,13,14,15);
      v8bf a1l = *(const v8bf*)&As[(wm + 16 + ml) * ALD + hk];
      v8bf a1h = *(const v8bf*)&As[(wm + 16 + ml) * ALD + hk + 16];
      v16bf a1 = __builtin_shufflevector(a1l, a1h,
          0,1,2,3,4,5,6,7,8,9,10,11,12,13,14,15);
      // B fragments: N = lane&15 within 16-wide subtile, K-half by lane>>4.
      v8bf b0l = *(const v8bf*)&Bs[(wn + ml) * BLD + kb];
      v8bf b0h = *(const v8bf*)&Bs[(wn + ml) * BLD + kb + 16];
      v16bf b0 = __builtin_shufflevector(b0l, b0h,
          0,1,2,3,4,5,6,7,8,9,10,11,12,13,14,15);
      v8bf b1l = *(const v8bf*)&Bs[(wn + 16 + ml) * BLD + kb];
      v8bf b1h = *(const v8bf*)&Bs[(wn + 16 + ml) * BLD + kb + 16];
      v16bf b1 = __builtin_shufflevector(b1l, b1h,
          0,1,2,3,4,5,6,7,8,9,10,11,12,13,14,15);
      acc00 = __builtin_amdgcn_wmma_f32_16x16x32_bf16(
          false, a0, false, b0, (short)0, acc00, false, false);
      acc01 = __builtin_amdgcn_wmma_f32_16x16x32_bf16(
          false, a0, false, b1, (short)0, acc01, false, false);
      acc10 = __builtin_amdgcn_wmma_f32_16x16x32_bf16(
          false, a1, false, b0, (short)0, acc10, false, false);
      acc11 = __builtin_amdgcn_wmma_f32_16x16x32_bf16(
          false, a1, false, b1, (short)0, acc11, false, false);
    }
    __syncthreads();
  }

  // C/D layout: VGPR r -> row (r) for lanes<16, row (8+r) for lanes>=16; col = lane&15.
  const int ml   = lane & 15;
  const int rofs = (lane >> 4) << 3;
  const int gc0  = n0 + wn + ml;
  const int gc1  = gc0 + 16;
  const float bias0 = bias ? bias[gc0] : 0.f;
  const float bias1 = bias ? bias[gc1] : 0.f;
  #pragma unroll
  for (int r = 0; r < 8; ++r) {
    int row0 = m0 + wm + rofs + r;
    int row1 = row0 + 16;
    float v00 = acc00[r] + bias0, v01 = acc01[r] + bias1;
    float v10 = acc10[r] + bias0, v11 = acc11[r] + bias1;
    if (addsrc) {
      v00 += addsrc[(size_t)row0 * N + gc0]; v01 += addsrc[(size_t)row0 * N + gc1];
      v10 += addsrc[(size_t)row1 * N + gc0]; v11 += addsrc[(size_t)row1 * N + gc1];
    }
    if (relu) {
      v00 = fmaxf(v00, 0.f); v01 = fmaxf(v01, 0.f);
      v10 = fmaxf(v10, 0.f); v11 = fmaxf(v11, 0.f);
    }
    C[(size_t)row0 * N + gc0] = v00; C[(size_t)row0 * N + gc1] = v01;
    C[(size_t)row1 * N + gc0] = v10; C[(size_t)row1 * N + gc1] = v11;
  }
}

// ---------------- GCN normalization helpers ----------------
__global__ void k_fill_deg(float* deg) {
  int v = blockIdx.x * blockDim.x + threadIdx.x;
  if (v < V_) deg[v] = 1.0f;                     // self-loop
}
__global__ void k_deg_acc(const int* __restrict__ ei, float* deg) {
  int e = blockIdx.x * blockDim.x + threadIdx.x;
  if (e < E_) atomicAdd(&deg[ei[E_ + e]], 1.0f); // cols = edge_index[1]
}
__global__ void k_dinv(float* d) {
  int v = blockIdx.x * blockDim.x + threadIdx.x;
  if (v < V_) d[v] = rsqrtf(d[v]);               // deg >= 1 always
}

// out = bias + tmp*dinv[v]^2 (self loop) (+ extra). grid.y = batch; F = 1<<fs.
__global__ void k_gcn_init(const float* __restrict__ tmp,
                           const float* __restrict__ bias,
                           const float* __restrict__ dinv,
                           const float* __restrict__ extra,
                           float* __restrict__ out, int fs) {
  int i = blockIdx.x * blockDim.x + threadIdx.x; // < V*F (exact multiple of 256)
  int nvf = V_ << fs;
  if (i >= nvf) return;
  int f = i & ((1 << fs) - 1);
  int v = i >> fs;
  int g = blockIdx.y * nvf + i;
  float dv = dinv[v];
  float val = bias[f] + tmp[g] * dv * dv;
  if (extra) val += extra[g];
  out[g] = val;
}

// out[b, dst, :] += tmp[b, src, :] * dinv[src]*dinv[dst]  (float4 + f32 atomics, L2-resident)
// grid.y = batch; F4 = F/4 = 1<<f4s (so 32 lanes of a wave cover one edge for F>=128).
__global__ void k_gcn_edges(const float* __restrict__ tmp,
                            const int* __restrict__ ei,
                            const float* __restrict__ dinv,
                            float* __restrict__ out, int f4s) {
  int r = blockIdx.x * blockDim.x + threadIdx.x;
  if (r >= (E_ << f4s)) return;
  int e  = r >> f4s;                             // wave-uniform for F>=128
  int f4 = r & ((1 << f4s) - 1);
  int F  = 4 << f4s;
  int src = ei[e];
  int dst = ei[E_ + e];
  float w = dinv[src] * dinv[dst];
  int base = blockIdx.y * (V_ << (f4s + 2));
  const float4 v = *(const float4*)&tmp[base + src * F + f4 * 4];
  float* o = &out[base + dst * F + f4 * 4];
  atomicAdd(o + 0, v.x * w);
  atomicAdd(o + 1, v.y * w);
  atomicAdd(o + 2, v.z * w);
  atomicAdd(o + 3, v.w * w);
}

__global__ void k_relu(float* z, int n) {
  int i = blockIdx.x * blockDim.x + threadIdx.x;
  if (i < n) z[i] = fmaxf(z[i], 0.f);
}

// gb[b, :] = t_enc[b] @ filmW + filmb   (tiny: M=4)
__global__ void k_film(const float* __restrict__ tenc,
                       const float* __restrict__ W,
                       const float* __restrict__ b,
                       float* __restrict__ gb, int N2) {
  int i = blockIdx.x * blockDim.x + threadIdx.x;
  if (i >= B_ * N2) return;
  int bb = i / N2, n = i - bb * N2;
  float s = b[n];
  #pragma unroll 4
  for (int k = 0; k < ENC_; ++k) s += tenc[bb * ENC_ + k] * W[(size_t)k * N2 + n];
  gb[i] = s;
}

// z = relu(z * gamma[b,c] + beta[b,c]); gamma = gb[:, :h], beta = gb[:, h:].
// grid.y = batch; h = 1<<hs.
__global__ void k_film_apply(float* __restrict__ z,
                             const float* __restrict__ gb, int hs) {
  int i = blockIdx.x * blockDim.x + threadIdx.x; // < V*h
  int nvh = V_ << hs;
  if (i >= nvh) return;
  int c = i & ((1 << hs) - 1);
  int bb = blockIdx.y;
  float g  = gb[(bb << (hs + 1)) + c];
  float be = gb[(bb << (hs + 1)) + (1 << hs) + c];
  int gidx = bb * nvh + i;
  z[gidx] = fmaxf(z[gidx] * g + be, 0.f);
}

// ---------------- host orchestration ----------------
extern "C" void kernel_launch(void* const* d_in, const int* in_sizes, int n_in,
                              void* d_out, int out_size, void* d_ws, size_t ws_size,
                              hipStream_t stream) {
  (void)in_sizes; (void)n_in; (void)out_size; (void)ws_size;
  const float* x_in = (const float*)d_in[0];     // [B,V,64]
  const float* cond = (const float*)d_in[1];     // [V,32]
  const float* tenc = (const float*)d_in[2];     // [B,128]
  const int*   ei   = (const int*)d_in[3];       // [2,E]
  auto P = [&](int i) { return (const float*)d_in[i]; };

  // JAX pytree (sorted-key) flatten order of params:
  // resblock leaves: film.W, film.b, g0.W, g0.b, g1.W, g1.b, gn.beta, gn.gamma,
  //                  l0.W, l0.b, l1.W, l1.b   (12 per block; gn is dead code)
  // down0: blocks@4,@16; in.W=28,in.b=29
  // down1: blocks@30,@42; in=54/55; out=56/57
  // up:    blocks@58,@70; in=82/83; out=84/85

  size_t off = 0;
  auto alloc = [&](size_t bytes) {
    size_t o = off; off += (bytes + 255) & ~(size_t)255;
    return (float*)((char*)d_ws + o);
  };
  float* dinv = alloc((size_t)V_ * 4);           // deg then dinv in place
  float* gb   = alloc((size_t)B_ * 512 * 4);
  float* Z0   = alloc((size_t)M_ * 128 * 4);     // skip (down0 output)
  float* Z1   = alloc((size_t)M_ * 256 * 4);
  float* T1   = alloc((size_t)M_ * 256 * 4);
  float* T2   = alloc((size_t)M_ * 256 * 4);
  float* T3   = alloc((size_t)M_ * 256 * 4);     // total ~369 MB

  auto gemm = [&](const float* A1, int lda1, int k1,
                  const float* A2, int lda2, int k2, int a2b,
                  const float* W, int N, const float* bias,
                  const float* addsrc, float* C, int relu) {
    dim3 g(M_ / BM, N / BN);
    k_gemm_bf16<<<g, 256, 0, stream>>>(A1, lda1, k1, A2, lda2, a2b,
                                       W, bias, addsrc, C, N, k1 + k2, relu);
  };
  auto gcn_agg = [&](const float* tmp, const float* bias, const float* extra,
                     float* out, int F, int relu) {
    int fs = __builtin_ctz(F);
    dim3 gi(((V_ << fs) + 255) / 256, B_);
    k_gcn_init<<<gi, 256, 0, stream>>>(tmp, bias, dinv, extra, out, fs);
    int f4s = fs - 2;
    dim3 ge(((E_ << f4s) + 255) / 256, B_);
    k_gcn_edges<<<ge, 256, 0, stream>>>(tmp, ei, dinv, out, f4s);
    if (relu) {
      int tot = B_ * (V_ << fs);
      k_relu<<<(tot + 255) / 256, 256, 0, stream>>>(out, tot);
    }
  };
  auto res_block = [&](float* z, int h, int pb, float* X, float* Y, float* Zb) {
    // Y = relu(gcn([z|cond], g0))
    gemm(z, h, h, cond, COND_, COND_, 1, P(pb + 2), h, nullptr, nullptr, X, 0);
    gcn_agg(X, P(pb + 3), nullptr, Y, h, 1);
    // Zb = relu(Y @ l0 + b)
    gemm(Y, h, h, nullptr, 0, 0, 0, P(pb + 8), h, P(pb + 9), nullptr, Zb, 1);
    // Y = gcn(Zb, g1)
    gemm(Zb, h, h, nullptr, 0, 0, 0, P(pb + 4), h, nullptr, nullptr, X, 0);
    gcn_agg(X, P(pb + 5), nullptr, Y, h, 0);
    // FiLM + relu (in place on Y)
    k_film<<<(B_ * 2 * h + 255) / 256, 256, 0, stream>>>(tenc, P(pb + 0), P(pb + 1), gb, 2 * h);
    int hs = __builtin_ctz(h);
    dim3 gf(((V_ << hs) + 255) / 256, B_);
    k_film_apply<<<gf, 256, 0, stream>>>(Y, gb, hs);
    // z = Y @ l1 + b + z_skip   (residual fused into writeback, in place)
    gemm(Y, h, h, nullptr, 0, 0, 0, P(pb + 10), h, P(pb + 11), z, z, 0);
  };

  // GCN symmetric normalization (self-loops included)
  k_fill_deg<<<(V_ + 255) / 256, 256, 0, stream>>>(dinv);
  k_deg_acc<<<(E_ + 255) / 256, 256, 0, stream>>>(ei, dinv);
  k_dinv<<<(V_ + 255) / 256, 256, 0, stream>>>(dinv);

  // down0.in : x(64) -> Z0(128)
  gemm(x_in, IN_, IN_, nullptr, 0, 0, 0, P(28), 128, nullptr, nullptr, T1, 0);
  gcn_agg(T1, P(29), nullptr, Z0, 128, 0);
  res_block(Z0, 128, 4,  T1, T2, T3);
  res_block(Z0, 128, 16, T1, T2, T3);            // Z0 = skip

  // down1.in : Z0(128) -> Z1(256)
  gemm(Z0, 128, 128, nullptr, 0, 0, 0, P(54), 256, nullptr, nullptr, T1, 0);
  gcn_agg(T1, P(55), nullptr, Z1, 256, 0);
  res_block(Z1, 256, 30, T1, T2, T3);
  res_block(Z1, 256, 42, T1, T2, T3);

  // down1.out : Z1(256) -> T2(128)
  gemm(Z1, 256, 256, nullptr, 0, 0, 0, P(56), 128, nullptr, nullptr, T1, 0);
  gcn_agg(T1, P(57), nullptr, T2, 128, 0);

  // up.in : [T2 | Z0](256) -> T3(128)   (concat never materialized)
  gemm(T2, 128, 128, Z0, 128, 128, 0, P(82), 128, nullptr, nullptr, T1, 0);
  gcn_agg(T1, P(83), nullptr, T3, 128, 0);
  res_block(T3, 128, 58, T1, T2, Z1);
  res_block(T3, 128, 70, T1, T2, Z1);

  // up.out : T3(128) -> 64, final residual x_in fused into aggregation init
  gemm(T3, 128, 128, nullptr, 0, 0, 0, P(84), 64, nullptr, nullptr, T1, 0);
  gcn_agg(T1, P(85), x_in, (float*)d_out, 64, 0);
}
